// Mamba2_18373870092677
// MI455X (gfx1250) — matrix operand order repaired and checked by reading, not compiled, not run on hardware
//
#include <hip/hip_runtime.h>
#include <hip/hip_bf16.h>

typedef __attribute__((ext_vector_type(16))) _Float16 v16h;
typedef __attribute__((ext_vector_type(8)))  float    v8f;

#define B_   2
#define L_   1024
#define DM   768
#define DI   1536
#define DS_  64
#define HD   8
#define NH   192
#define CD   1664
#define DIP  3392
#define CH   128
#define NC   8

// workspace layout (in floats)
#define OFF_ZX  ((size_t)0)
#define SZ_ZX   ((size_t)B_*L_*DIP)
#define OFF_XBC (OFF_ZX + SZ_ZX)
#define SZ_XBC  ((size_t)B_*L_*CD)
#define OFF_DT  (OFF_XBC + SZ_XBC)
#define SZ_DT   ((size_t)B_*L_*NH)
#define OFF_ACS (OFF_DT + SZ_DT)
#define SZ_ACS  ((size_t)B_*NC*NH*CH)
#define OFF_SC  (OFF_ACS + SZ_ACS)              /* _Float16 region */
#define SZ_SC_F ((size_t)(B_*NC*CH*CH)/2)
#define OFF_ST  (OFF_SC + SZ_SC_F)
#define SZ_ST   ((size_t)B_*NC*NH*HD*DS_)
#define OFF_Y   (OFF_ST + SZ_ST)

// ---- WMMA helpers (CDNA5 16x16x32 f16 -> f32) ----
__device__ __forceinline__ int kmapA(int e, int lane) {
  // 16-bit A 16x32 layout: V0..3 -> K0..7 (lanes<16) / K8..15 (lanes>=16); V4..7 -> +16
  return (e & 7) + ((lane & 16) ? 8 : 0) + ((e & 8) ? 16 : 0);
}
__device__ __forceinline__ int kmapB(int e, int lane) {
  // 16-bit B 32x16 layout: same K order as A: V0..3 -> K0..7 (lanes<16) / K8..15 (lanes>=16); V4..7 -> +16
  return (e & 7) + ((lane & 16) ? 8 : 0) + ((e & 8) ? 16 : 0);
}
__device__ __forceinline__ v8f wmma32(v16h a, v16h b, v8f c) {
  return __builtin_amdgcn_wmma_f32_16x16x32_f16(false, a, false, b, (short)0, c, false, false);
}
__device__ __forceinline__ v8f v8f_zero() {
  v8f z = {0.f,0.f,0.f,0.f,0.f,0.f,0.f,0.f};
  return z;
}

// ================= TDM demo: stage a 32x16 f32 tile of W_in into LDS =================
#if defined(__has_builtin)
#if __has_builtin(__builtin_amdgcn_tensor_load_to_lds) && __has_builtin(__builtin_amdgcn_s_wait_tensorcnt)
#define HAVE_TDM 1
#endif
#endif
#ifndef HAVE_TDM
#define HAVE_TDM 0
#endif

#if HAVE_TDM
typedef __attribute__((ext_vector_type(4))) unsigned int u32x4;
typedef __attribute__((ext_vector_type(4))) int          i32x4;
typedef __attribute__((ext_vector_type(8))) int          i32x8;

__global__ void k_tdm_prefetch(const float* __restrict__ Wi, float* __restrict__ tok) {
  __shared__ float tile[32 * 16];
  unsigned long long ga = (unsigned long long)(const void*)Wi;
  unsigned int lds_addr = (unsigned int)(unsigned long long)(void*)&tile[0];
  u32x4 g0;
  g0.x = 1u;                                              // count=1, user descriptor
  g0.y = lds_addr;                                        // lds_addr
  g0.z = (unsigned int)(ga & 0xffffffffu);                // global_addr[31:0]
  g0.w = (unsigned int)((ga >> 32) & 0x01ffffffu) | (2u << 30);  // addr[56:32] | type=2
  i32x8 g1;
  g1[0] = (int)(2u << 16);                                // data_size = 4B
  g1[1] = (int)((DM & 0xffff) << 16);                     // tensor_dim0 lo16
  g1[2] = (int)((DM >> 16) | ((DIP & 0xffff) << 16));     // dim0 hi16 | tensor_dim1 lo16
  g1[3] = (int)((DIP >> 16) | (32 << 16));                // dim1 hi16 | tile_dim0 = 32
  g1[4] = (int)16;                                        // tile_dim1 = 16, tile_dim2 = 0
  g1[5] = (int)DM;                                        // tensor_dim0_stride lo32
  g1[6] = 0;
  g1[7] = 0;
  i32x4 gz = {0, 0, 0, 0};
#if __clang_major__ >= 23
  i32x8 gz8 = {0, 0, 0, 0, 0, 0, 0, 0};
  __builtin_amdgcn_tensor_load_to_lds(g0, g1, gz, gz, gz8, 0);
#else
  __builtin_amdgcn_tensor_load_to_lds(g0, g1, gz, gz, 0);
#endif
  __builtin_amdgcn_s_wait_tensorcnt(0);
  __syncthreads();
  if (threadIdx.x == 0) tok[0] = tile[0];  // token overwritten later by k_ydiag
}
#endif

// ================= 1) in-projection GEMM: zx[m,n] = sum_k x[b,k,l] * W_in[n,k] =================
__global__ void k_inproj(const float* __restrict__ x, const float* __restrict__ Wi,
                         float* __restrict__ zx) {
  const int lane = threadIdx.x & 31, wave = threadIdx.x >> 5;
  int wg = blockIdx.x * 8 + wave;
  const int MT = (B_ * L_) / 16, NT = DIP / 16;
  if (wg >= MT * NT) return;
  int tm = wg % MT, tn = wg / MT;
  int m0 = tm * 16, n0 = tn * 16;
  int mrow = m0 + (lane & 15);
  int bb = mrow >> 10, ll = mrow & (L_ - 1);
  int ncol = n0 + (lane & 15);
  v8f acc = v8f_zero();
  for (int k0 = 0; k0 < DM; k0 += 32) {
    __builtin_prefetch(&Wi[(size_t)ncol * DM + k0 + 32], 0, 3);
    v16h af, bf;
#pragma unroll
    for (int e = 0; e < 16; ++e) {
      int ka = k0 + kmapA(e, lane);
      int kb = k0 + kmapB(e, lane);
      af[e] = (_Float16)x[((size_t)bb * DM + ka) * L_ + ll];   // u[m][k] = x[b][k][l]
      bf[e] = (_Float16)Wi[(size_t)ncol * DM + kb];            // B[k][n] = W_in[n][k]
    }
    acc = wmma32(af, bf, acc);
  }
  int rb = m0 + ((lane & 16) ? 8 : 0);
#pragma unroll
  for (int r = 0; r < 8; ++r)
    zx[(size_t)(rb + r) * DIP + ncol] = acc[r];
}

// ================= 2) depthwise causal conv(4) + bias + SiLU over xBC channels =================
__global__ void k_convsilu(const float* __restrict__ zx, const float* __restrict__ cw,
                           const float* __restrict__ cb, float* __restrict__ xbc) {
  size_t idx = (size_t)blockIdx.x * blockDim.x + threadIdx.x;
  if (idx >= (size_t)B_ * L_ * CD) return;
  int c = (int)(idx % CD);
  int ml = (int)(idx / CD);
  int l = ml & (L_ - 1);
  float acc = cb[c];
#pragma unroll
  for (int j = 0; j < 4; ++j) {
    int ls = l + j - 3;
    if (ls >= 0)
      acc += zx[(size_t)(ml - l + ls) * DIP + DI + c] * cw[c * 4 + j];
  }
  float s = 1.f / (1.f + __expf(-acc));
  xbc[idx] = acc * s;
}

// ================= 3) dt = softplus(zx[..., -NH:] + dt_bias) =================
__global__ void k_dt(const float* __restrict__ zx, const float* __restrict__ dtb,
                     float* __restrict__ dt) {
  int idx = blockIdx.x * blockDim.x + threadIdx.x;
  if (idx >= B_ * L_ * NH) return;
  int h = idx % NH, ml = idx / NH;
  float v = zx[(size_t)ml * DIP + (DIP - NH) + h] + dtb[h];
  dt[idx] = (v > 20.f) ? v : log1pf(__expf(v));
}

// ================= 4) per-(b,h,chunk) cumsum of dA = dt * (-exp(A_log)) =================
__global__ void k_acs(const float* __restrict__ dt, const float* __restrict__ Alog,
                      float* __restrict__ acs) {
  int idx = blockIdx.x * blockDim.x + threadIdx.x;
  if (idx >= B_ * NH * NC) return;
  int c = idx % NC, h = (idx / NC) % NH, b = idx / (NC * NH);
  float A = -__expf(Alog[h]);
  float s = 0.f;
  size_t ob = ((size_t)(b * NC + c) * NH + h) * CH;
  for (int i = 0; i < CH; ++i) {
    int l = c * CH + i;
    s += dt[((size_t)b * L_ + l) * NH + h] * A;
    acs[ob + i] = s;
  }
}

// ================= 5) scores[b,c,l,s] = sum_n Cm[l,n] * Bm[s,n]  (stored f16) =================
__global__ void k_scores(const float* __restrict__ xbc, _Float16* __restrict__ sc) {
  const int lane = threadIdx.x & 31, wave = threadIdx.x >> 5;
  int wg = blockIdx.x * 8 + wave;  // 1024 waves
  int t = wg & 63, bc = wg >> 6;
  int b = bc >> 3, c = bc & 7;
  int tm = t >> 3, tn = t & 7;
  int lrow = tm * 16 + (lane & 15);
  int scol = tn * 16 + (lane & 15);
  size_t rowC = ((size_t)b * L_ + c * CH + lrow) * CD + DI + DS_;  // Cm at 1600
  size_t rowB = ((size_t)b * L_ + c * CH + scol) * CD + DI;       // Bm at 1536
  v8f acc = v8f_zero();
  for (int k0 = 0; k0 < DS_; k0 += 32) {
    v16h af, bf;
#pragma unroll
    for (int e = 0; e < 16; ++e) {
      af[e] = (_Float16)xbc[rowC + k0 + kmapA(e, lane)];
      bf[e] = (_Float16)xbc[rowB + k0 + kmapB(e, lane)];
    }
    acc = wmma32(af, bf, acc);
  }
  size_t ob = (size_t)bc * CH * CH;
  int rb = tm * 16 + ((lane & 16) ? 8 : 0);
#pragma unroll
  for (int r = 0; r < 8; ++r)
    sc[ob + (size_t)(rb + r) * CH + scol] = (_Float16)acc[r];
}

// ================= 6) Y_diag per head: ((scores .* L_h) @ xdt_h) + D*xh =================
__global__ void k_ydiag(const float* __restrict__ xbc, const float* __restrict__ dt,
                        const float* __restrict__ acs, const _Float16* __restrict__ sc,
                        const float* __restrict__ Dv, float* __restrict__ y) {
  __shared__ _Float16 s_sc[CH * CH];   // 32 KB
  __shared__ float s_acs[8][CH];       // 4 KB
  const int tid = threadIdx.x, lane = tid & 31, wave = tid >> 5;
  int bc = blockIdx.x / 24, g = blockIdx.x % 24;
  int b = bc >> 3, c = bc & 7;
  int h = g * 8 + wave;
  const _Float16* scg = sc + (size_t)bc * CH * CH;
  for (int i = tid; i < CH * CH; i += 256) s_sc[i] = scg[i];
  size_t ab = ((size_t)bc * NH + h) * CH;
  for (int i = lane; i < CH; i += 32) s_acs[wave][i] = acs[ab + i];
  __syncthreads();

  int p = lane & 15;
  v8f acc[8];
#pragma unroll
  for (int mt = 0; mt < 8; ++mt) acc[mt] = v8f_zero();
  for (int k0 = 0; k0 < CH; k0 += 32) {
    v16h bf;
#pragma unroll
    for (int e = 0; e < 16; ++e) {
      int s = k0 + kmapB(e, lane);
      float v = 0.f;
      if (p < 8) {
        size_t xi = ((size_t)b * L_ + c * CH + s) * CD + h * HD + p;
        v = xbc[xi] * dt[((size_t)b * L_ + c * CH + s) * NH + h];  // xdt[s,h,p]
      }
      bf[e] = (_Float16)v;
    }
#pragma unroll
    for (int mt = 0; mt < 8; ++mt) {
      int l = mt * 16 + (lane & 15);
      float acl = s_acs[wave][l];
      v16h af;
#pragma unroll
      for (int e = 0; e < 16; ++e) {
        int s = k0 + kmapA(e, lane);
        float v = 0.f;
        if (s <= l) v = (float)s_sc[l * CH + s] * __expf(acl - s_acs[wave][s]);
        af[e] = (_Float16)v;
      }
      acc[mt] = wmma32(af, bf, acc[mt]);
    }
  }
  if (p < 8) {
    float Dh = Dv[h];
#pragma unroll
    for (int mt = 0; mt < 8; ++mt) {
      int rb = mt * 16 + ((lane & 16) ? 8 : 0);
#pragma unroll
      for (int r = 0; r < 8; ++r) {
        int l = c * CH + rb + r;
        size_t xi = ((size_t)b * L_ + l) * CD + h * HD + p;
        y[(((size_t)b * L_ + l) * NH + h) * HD + p] = acc[mt][r] + Dh * xbc[xi];
      }
    }
  }
}

// ================= 7) per-chunk states: st[b,c,h,p,n] = sum_l xdt[l,p]*decay[l]*Bm[l,n] =================
__global__ void k_states(const float* __restrict__ xbc, const float* __restrict__ dt,
                         const float* __restrict__ acs, float* __restrict__ st) {
  const int lane = threadIdx.x & 31, wave = threadIdx.x >> 5;
  int wg = blockIdx.x * 8 + wave;  // 3072 waves = (b*NC+c)*NH+h
  int h = wg % NH, c = (wg / NH) & 7, b = wg / (NH * NC);
  size_t ab = ((size_t)(b * NC + c) * NH + h) * CH;
  float alast = acs[ab + CH - 1];
  int p = lane & 15;
  v8f acc[4];
#pragma unroll
  for (int nt = 0; nt < 4; ++nt) acc[nt] = v8f_zero();
  for (int k0 = 0; k0 < CH; k0 += 32) {
    v16h af;
#pragma unroll
    for (int e = 0; e < 16; ++e) {
      int l = k0 + kmapA(e, lane);
      float v = 0.f;
      if (p < 8) {
        size_t xi = ((size_t)b * L_ + c * CH + l) * CD + h * HD + p;
        v = xbc[xi] * dt[((size_t)b * L_ + c * CH + l) * NH + h] * __expf(alast - acs[ab + l]);
      }
      af[e] = (_Float16)v;
    }
#pragma unroll
    for (int nt = 0; nt < 4; ++nt) {
      v16h bf;
#pragma unroll
      for (int e = 0; e < 16; ++e) {
        int l = k0 + kmapB(e, lane);
        int n = nt * 16 + (lane & 15);
        bf[e] = (_Float16)xbc[((size_t)b * L_ + c * CH + l) * CD + DI + n];
      }
      acc[nt] = wmma32(af, bf, acc[nt]);
    }
  }
  if (!(lane & 16)) {  // rows 0..7 hold valid p
    size_t ob = (size_t)wg * HD * DS_;
#pragma unroll
    for (int nt = 0; nt < 4; ++nt)
#pragma unroll
      for (int r = 0; r < 8; ++r)
        st[ob + (size_t)r * DS_ + nt * 16 + (lane & 15)] = acc[nt][r];
  }
}

// ================= 8) inter-chunk scan: st[b,c] := sum_{c'<c} decay * st[b,c'] (prev states) =================
__global__ void k_scan(const float* __restrict__ acs, float* __restrict__ st) {
  int idx = blockIdx.x * blockDim.x + threadIdx.x;
  if (idx >= B_ * NH * HD * DS_) return;
  int n = idx & 63, p = (idx >> 6) & 7, h = (idx >> 9) % NH, b = idx / (DS_ * HD * NH);
  float S = 0.f;
  for (int c = 0; c < NC; ++c) {
    size_t si = (((size_t)(b * NC + c) * NH + h) * HD + p) * DS_ + n;
    float v = st[si];
    st[si] = S;  // becomes "prev" state entering chunk c
    float dec = __expf(acs[((size_t)(b * NC + c) * NH + h) * CH + CH - 1]);
    S = S * dec + v;
  }
}

// ================= 9) Y_off: y += (Cm .* exp(Acs[l])) @ prev^T =================
__global__ void k_yoff(const float* __restrict__ xbc, const float* __restrict__ acs,
                       const float* __restrict__ st, float* __restrict__ y) {
  const int lane = threadIdx.x & 31, wave = threadIdx.x >> 5;
  int wg = blockIdx.x * 8 + wave;
  int h = wg % NH, c = (wg / NH) & 7, b = wg / (NH * NC);
  size_t ab = ((size_t)(b * NC + c) * NH + h) * CH;
  size_t sb = ((size_t)(b * NC + c) * NH + h) * HD * DS_;
  int p = lane & 15;
  v8f acc[8];
#pragma unroll
  for (int mt = 0; mt < 8; ++mt) acc[mt] = v8f_zero();
  for (int k0 = 0; k0 < DS_; k0 += 32) {
    v16h bf;
#pragma unroll
    for (int e = 0; e < 16; ++e) {
      int n = k0 + kmapB(e, lane);
      bf[e] = (_Float16)((p < 8) ? st[sb + (size_t)p * DS_ + n] : 0.f);
    }
#pragma unroll
    for (int mt = 0; mt < 8; ++mt) {
      int lr = mt * 16 + (lane & 15);
      float ecl = __expf(acs[ab + lr]);
      v16h af;
#pragma unroll
      for (int e = 0; e < 16; ++e) {
        int n = k0 + kmapA(e, lane);
        af[e] = (_Float16)(xbc[((size_t)b * L_ + c * CH + lr) * CD + DI + DS_ + n] * ecl);
      }
      acc[mt] = wmma32(af, bf, acc[mt]);
    }
  }
  if (p < 8) {
#pragma unroll
    for (int mt = 0; mt < 8; ++mt) {
      int rb = mt * 16 + ((lane & 16) ? 8 : 0);
#pragma unroll
      for (int r = 0; r < 8; ++r) {
        int l = c * CH + rb + r;
        y[(((size_t)b * L_ + l) * NH + h) * HD + p] += acc[mt][r];
      }
    }
  }
}

// ================= 10) gate with silu(z), RMS norm * norm_w =================
__global__ void k_gatenorm(const float* __restrict__ zx, const float* __restrict__ nw,
                           float* __restrict__ y) {
  __shared__ float red[256];
  int ml = blockIdx.x;
  int tid = threadIdx.x;
  float loc[6];
  float ss = 0.f;
#pragma unroll
  for (int j = 0; j < 6; ++j) {
    int i = j * 256 + tid;
    float z = zx[(size_t)ml * DIP + i];
    float sig = 1.f / (1.f + __expf(-z));
    float v = y[(size_t)ml * DI + i] * z * sig;
    loc[j] = v;
    ss += v * v;
  }
  red[tid] = ss;
  __syncthreads();
  for (int o = 128; o > 0; o >>= 1) {
    if (tid < o) red[tid] += red[tid + o];
    __syncthreads();
  }
  float scl = rsqrtf(red[0] / (float)DI + 1e-5f);
#pragma unroll
  for (int j = 0; j < 6; ++j) {
    int i = j * 256 + tid;
    y[(size_t)ml * DI + i] = loc[j] * scl * nw[i];
  }
}

// ================= 11) out-projection GEMM, store transposed (B, DM, L) =================
__global__ void k_outproj(const float* __restrict__ y, const float* __restrict__ Wo,
                          float* __restrict__ out) {
  const int lane = threadIdx.x & 31, wave = threadIdx.x >> 5;
  int wg = blockIdx.x * 8 + wave;  // 6144 tiles
  const int MT = (B_ * L_) / 16;
  int tm = wg % MT, tn = wg / MT;
  int m0 = tm * 16, n0 = tn * 16;
  int mrow = m0 + (lane & 15);
  int ncol = n0 + (lane & 15);
  v8f acc = v8f_zero();
  for (int k0 = 0; k0 < DI; k0 += 32) {
    __builtin_prefetch(&Wo[(size_t)ncol * DI + k0 + 32], 0, 3);
    v16h af, bf;
#pragma unroll
    for (int e = 0; e < 16; ++e) {
      af[e] = (_Float16)y[(size_t)mrow * DI + k0 + kmapA(e, lane)];
      bf[e] = (_Float16)Wo[(size_t)ncol * DI + k0 + kmapB(e, lane)];
    }
    acc = wmma32(af, bf, acc);
  }
#pragma unroll
  for (int r = 0; r < 8; ++r) {
    int m = m0 + r + ((lane & 16) ? 8 : 0);
    int b = m >> 10, l = m & (L_ - 1);
    out[((size_t)b * DM + ncol) * L_ + l] = acc[r];
  }
}

extern "C" void kernel_launch(void* const* d_in, const int* in_sizes, int n_in,
                              void* d_out, int out_size, void* d_ws, size_t ws_size,
                              hipStream_t stream) {
  (void)in_sizes; (void)n_in; (void)out_size; (void)ws_size;
  const float* x    = (const float*)d_in[0];
  const float* W_in = (const float*)d_in[1];
  const float* cw   = (const float*)d_in[2];
  const float* cb   = (const float*)d_in[3];
  const float* dtb  = (const float*)d_in[4];
  const float* Alog = (const float*)d_in[5];
  const float* Dv   = (const float*)d_in[6];
  const float* nw   = (const float*)d_in[7];
  const float* W_out = (const float*)d_in[8];
  float* out = (float*)d_out;
  float* ws = (float*)d_ws;
  float*    zx  = ws + OFF_ZX;
  float*    xbc = ws + OFF_XBC;
  float*    dt  = ws + OFF_DT;
  float*    acs = ws + OFF_ACS;
  _Float16* sc  = (_Float16*)(ws + OFF_SC);
  float*    st  = ws + OFF_ST;
  float*    y   = ws + OFF_Y;

#if HAVE_TDM
  k_tdm_prefetch<<<1, 32, 0, stream>>>(W_in, y);  // token slot overwritten by k_ydiag
#endif
  k_inproj<<<3392, 256, 0, stream>>>(x, W_in, zx);
  k_convsilu<<<(B_ * L_ * CD + 255) / 256, 256, 0, stream>>>(zx, cw, cb, xbc);
  k_dt<<<(B_ * L_ * NH + 255) / 256, 256, 0, stream>>>(zx, dtb, dt);
  k_acs<<<(B_ * NH * NC + 255) / 256, 256, 0, stream>>>(dt, Alog, acs);
  k_scores<<<128, 256, 0, stream>>>(xbc, sc);
  k_ydiag<<<384, 256, 0, stream>>>(xbc, dt, acs, sc, Dv, y);
  k_states<<<384, 256, 0, stream>>>(xbc, dt, acs, st);
  k_scan<<<(B_ * NH * HD * DS_ + 255) / 256, 256, 0, stream>>>(acs, st);
  k_yoff<<<384, 256, 0, stream>>>(xbc, acs, st, y);
  k_gatenorm<<<B_ * L_, 256, 0, stream>>>(zx, nw, y);
  k_outproj<<<768, 256, 0, stream>>>(y, W_out, out);
}
